// SpikingBlock_5729486373026
// MI455X (gfx1250) — matrix operand order repaired and checked by reading, not compiled, not run on hardware
//
#include <hip/hip_runtime.h>

// ---------------------------------------------------------------------------
// LIF spiking layer, fused: out[t] = spike(u); u = 0.75u + x_t@W + b - out
// GEMM via V_WMMA_F32_16X16X32_BF16 with a 2-term bf16 split of both operands
// (hi*hi + hi*lo + lo*hi ~ fp32 accuracy). W tile lives in LDS (320KB/WGP),
// padded to a conflict-free stride; A streams from global (L2-resident).
// ---------------------------------------------------------------------------

#define T_DIM 128
#define B_DIM 128
#define K_DIM 1024
#define N_DIM 1024
// per-t stride in elements for X / out: B_DIM * N_DIM(=K_DIM) = 1<<17
#define TSTRIDE (1u << 17)
// LDS row stride in shorts: 1032 shorts = 516 dwords == 4 (mod 64 banks)
#define LPAD 1032

typedef __attribute__((ext_vector_type(16))) __bf16 v16bf;
typedef __attribute__((ext_vector_type(8)))  float  v8f;

union Frag {
  v16bf v;
  uint4 q[2];
  unsigned short u[16];
};

// Split f32 into bf16 hi (truncated) + bf16 lo (residual). Pure bit ops so we
// never rely on __bf16 scalar arithmetic support.
__device__ __forceinline__ void split1(float f, unsigned short& h, unsigned short& l) {
  unsigned int bits = __float_as_uint(f);
  h = (unsigned short)(bits >> 16);
  float r = f - __uint_as_float(bits & 0xffff0000u);
  l = (unsigned short)(__float_as_uint(r) >> 16);
}

// ---- Prep: transpose W [K][N] -> Wt [N][K], split into bf16 hi/lo ----------
__global__ void split_w_kernel(const float* __restrict__ W,
                               unsigned short* __restrict__ Whi,
                               unsigned short* __restrict__ Wlo) {
  unsigned int idx = blockIdx.x * blockDim.x + threadIdx.x; // over N*K = 1M
  unsigned int n = idx >> 10;
  unsigned int k = idx & 1023u;
  float w = W[(size_t)k * N_DIM + n];   // transpose read (one-shot, L2 soaks it)
  unsigned short h, l;
  split1(w, h, l);
  Whi[idx] = h;
  Wlo[idx] = l;
}

// ---- Prep: elementwise split of X (row-major already matches A layout) -----
__global__ void split_x_kernel(const float* __restrict__ X,
                               unsigned short* __restrict__ Xhi,
                               unsigned short* __restrict__ Xlo) {
  unsigned int idx = blockIdx.x * blockDim.x + threadIdx.x; // over total/4
  float4 v = ((const float4*)X)[idx];
  ushort4 h, l;
  split1(v.x, h.x, l.x);
  split1(v.y, h.y, l.y);
  split1(v.z, h.z, l.z);
  split1(v.w, h.w, l.w);
  ((ushort4*)Xhi)[idx] = h;
  ((ushort4*)Xlo)[idx] = l;
}

// ---- Main fused kernel ------------------------------------------------------
// Block = 2 waves sharing one 16-column N-tile (W tile cached once in LDS);
// each wave owns a 16x16 (batch x out) tile and carries u in registers.
template <bool XSPLIT>
__global__ __launch_bounds__(64) void lif_wmma_kernel(
    const float* __restrict__ Xf,
    const unsigned short* __restrict__ Xhi,
    const unsigned short* __restrict__ Xlo,
    const unsigned short* __restrict__ Whi,
    const unsigned short* __restrict__ Wlo,
    const float* __restrict__ bias,
    float* __restrict__ out) {
  __shared__ __align__(16) unsigned short sWhi[16 * LPAD];
  __shared__ __align__(16) unsigned short sWlo[16 * LPAD];

  const int tid    = threadIdx.x;
  const int lane   = tid & 31;
  const int waveId = tid >> 5;                 // 0..1
  const int row    = lane & 15;
  const int half   = lane >> 4;                // 0 / 1
  const int tileN  = (blockIdx.x & 63) << 4;   // 64 N-tiles
  const int tileM  = ((((int)blockIdx.x >> 6) << 1) + waveId) << 4; // 8 M-tiles

  // Cooperative load of the W tile (16 cols x 1024 K, hi+lo) into LDS.
  // Padded stride => row r starts at bank 4r: conflict-free b128 reads.
  for (int i = tid; i < 16 * (K_DIM / 8); i += 64) {
    int col = i >> 7;             // 0..15
    int kc  = (i & 127) << 3;     // K chunk base (multiple of 8 shorts = 16B)
    *(uint4*)&sWhi[col * LPAD + kc] =
        *(const uint4*)(Whi + (((size_t)(tileN + col)) << 10) + kc);
    *(uint4*)&sWlo[col * LPAD + kc] =
        *(const uint4*)(Wlo + (((size_t)(tileN + col)) << 10) + kc);
  }
  __syncthreads();

  // B fragment (32x16, 16-bit): col = lane&15; elems 0-7 = K(k0 + 8*half ..),
  // 8-15 = +16 (matches A).
  const unsigned short* bh = sWhi + row * LPAD + (half << 3);
  const unsigned short* bl = sWlo + row * LPAD + (half << 3);

  // A fragment: row M = lane&15; elems 0-7 = K(k0 + 8*half ..), 8-15 = +16.
  const size_t arow = ((size_t)(tileM + row)) << 10;
  const int koff = half << 3;

  const float bv = bias[tileN + row];          // C layout: whole lane shares N
  float* orow = out + ((size_t)(tileM + (half << 3))) * N_DIM + tileN + row;

  v8f u = (v8f)0.0f;                           // membrane potential state

  for (int t = 0; t < T_DIM; ++t) {
    v8f c = (v8f)0.0f;
    const size_t tbase = (size_t)t * TSTRIDE;

    // Prefetch next timestep's A rows (global_prefetch_b8); overlaps the
    // recurrence epilogue + next k-loop ramp with the compulsory misses.
    if (t + 1 < T_DIM) {
      if constexpr (XSPLIT) {
        const unsigned short* nh = Xhi + tbase + TSTRIDE + arow;
        const unsigned short* nl = Xlo + tbase + TSTRIDE + arow;
        __builtin_prefetch(nh, 0, 3);
        __builtin_prefetch(nh + 1024, 0, 3);
        __builtin_prefetch(nl, 0, 3);
        __builtin_prefetch(nl + 1024, 0, 3);
      } else {
        const float* nf = Xf + tbase + TSTRIDE + arow;
        __builtin_prefetch(nf, 0, 3);
        __builtin_prefetch(nf + 512, 0, 3);
      }
    }

#pragma unroll 2
    for (int k0 = 0; k0 < K_DIM; k0 += 32) {
      Frag Ah, Al, Bh, Bl;
      Bh.q[0] = *(const uint4*)(bh + k0);
      Bh.q[1] = *(const uint4*)(bh + k0 + 16);
      Bl.q[0] = *(const uint4*)(bl + k0);
      Bl.q[1] = *(const uint4*)(bl + k0 + 16);

      if constexpr (XSPLIT) {
        const unsigned short* ah = Xhi + tbase + arow;
        const unsigned short* al = Xlo + tbase + arow;
        Ah.q[0] = *(const uint4*)(ah + k0 + koff);
        Ah.q[1] = *(const uint4*)(ah + k0 + 16 + koff);
        Al.q[0] = *(const uint4*)(al + k0 + koff);
        Al.q[1] = *(const uint4*)(al + k0 + 16 + koff);
      } else {
        const float* af = Xf + tbase + arow;
        float4 f0 = *(const float4*)(af + k0 + koff);
        float4 f1 = *(const float4*)(af + k0 + koff + 4);
        float4 f2 = *(const float4*)(af + k0 + 16 + koff);
        float4 f3 = *(const float4*)(af + k0 + 16 + koff + 4);
        split1(f0.x, Ah.u[0], Al.u[0]);   split1(f0.y, Ah.u[1], Al.u[1]);
        split1(f0.z, Ah.u[2], Al.u[2]);   split1(f0.w, Ah.u[3], Al.u[3]);
        split1(f1.x, Ah.u[4], Al.u[4]);   split1(f1.y, Ah.u[5], Al.u[5]);
        split1(f1.z, Ah.u[6], Al.u[6]);   split1(f1.w, Ah.u[7], Al.u[7]);
        split1(f2.x, Ah.u[8], Al.u[8]);   split1(f2.y, Ah.u[9], Al.u[9]);
        split1(f2.z, Ah.u[10], Al.u[10]); split1(f2.w, Ah.u[11], Al.u[11]);
        split1(f3.x, Ah.u[12], Al.u[12]); split1(f3.y, Ah.u[13], Al.u[13]);
        split1(f3.z, Ah.u[14], Al.u[14]); split1(f3.w, Ah.u[15], Al.u[15]);
      }

      // x@W ~= xh*Wh + xh*Wl + xl*Wh  (dropped lo*lo term: < 2^-14 relative)
      c = __builtin_amdgcn_wmma_f32_16x16x32_bf16(false, Ah.v, false, Bh.v,
                                                  (short)0, c, false, false);
      c = __builtin_amdgcn_wmma_f32_16x16x32_bf16(false, Ah.v, false, Bl.v,
                                                  (short)0, c, false, false);
      c = __builtin_amdgcn_wmma_f32_16x16x32_bf16(false, Al.v, false, Bh.v,
                                                  (short)0, c, false, false);
    }

    // LIF recurrence + spike + soft reset (C layout: M = tileM + e + 8*half)
    float* ot = orow + tbase;
#pragma unroll
    for (int e = 0; e < 8; ++e) {
      float uu = 0.75f * u[e] + c[e] + bv;
      float s = (uu >= 1.0f) ? 1.0f : 0.0f;
      ot[(size_t)e << 10] = s;
      u[e] = uu - s;
    }
  }
}

extern "C" void kernel_launch(void* const* d_in, const int* in_sizes, int n_in,
                              void* d_out, int out_size, void* d_ws, size_t ws_size,
                              hipStream_t stream) {
  const float* X  = (const float*)d_in[0];   // [128,128,1024] f32
  const float* W  = (const float*)d_in[1];   // [1024,1024] f32
  const float* bv = (const float*)d_in[2];   // [1024] f32
  float* out = (float*)d_out;                // [128,128,1024] f32 spikes

  unsigned short* Whi = (unsigned short*)d_ws;           // 2 MB
  unsigned short* Wlo = Whi + (size_t)N_DIM * K_DIM;     // 2 MB

  // Transpose + split W into bf16 hi/lo (one-shot, 4 MB, L2-resident after).
  split_w_kernel<<<(N_DIM * K_DIM) / 256, 256, 0, stream>>>(W, Whi, Wlo);

  const size_t xElems = (size_t)T_DIM * B_DIM * K_DIM;   // 16.7M
  const size_t needX  = (size_t)4 * (1u << 20) + 2 * xElems * sizeof(unsigned short);

  if (ws_size >= needX) {
    // Pre-split X too: hot loop is then pure b128/ds loads + WMMA.
    unsigned short* Xhi = Wlo + (size_t)N_DIM * K_DIM;
    unsigned short* Xlo = Xhi + xElems;
    split_x_kernel<<<(unsigned)(xElems / 4 / 256), 256, 0, stream>>>(X, Xhi, Xlo);
    lif_wmma_kernel<true><<<256, 64, 0, stream>>>(nullptr, Xhi, Xlo, Whi, Wlo, bv, out);
  } else {
    lif_wmma_kernel<false><<<256, 64, 0, stream>>>(X, nullptr, nullptr, Whi, Wlo, bv, out);
  }
}
